// PointNet2Decoder_83897891160262
// MI455X (gfx1250) — matrix operand order repaired and checked by reading, not compiled, not run on hardware
//
#include <hip/hip_runtime.h>
#include <hip/hip_bf16.h>

typedef _Float16 v16h __attribute__((ext_vector_type(16)));
typedef _Float16 v8h  __attribute__((ext_vector_type(8)));
typedef float    v8f  __attribute__((ext_vector_type(8)));
typedef int      v4i  __attribute__((ext_vector_type(4)));

#define BN_EPS 1e-3f

// Async global->LDS path (gfx1250). Guarded so both device and host passes,
// and both toolchains, compile regardless of builtin availability.
#if defined(__HIP_DEVICE_COMPILE__)
#if __has_builtin(__builtin_amdgcn_global_load_async_to_lds_b128)
#define ATHENA_ASYNC_LDS 1
#endif
#endif

// ---------------------------------------------------------------------------
// src W is row-major [K][N] fp32; dst Wt is row-major [N][K] f16
// ---------------------------------------------------------------------------
__global__ void transpose_f32_to_f16(const float* __restrict__ W,
                                     _Float16* __restrict__ Wt, int K, int N) {
  int idx = blockIdx.x * blockDim.x + threadIdx.x;
  if (idx < K * N) {
    int n = idx / K, k = idx - n * K;
    Wt[idx] = (_Float16)W[k * N + n];
  }
}

// ---------------------------------------------------------------------------
// deterministic BN finalize: reduce per-block partials in fixed order, fold
// (gamma, beta, mean, var) into out = a*z + c
// partials layout: [block][ch] = sum (ch<128), [block][128+ch] = sumsq
// ---------------------------------------------------------------------------
__global__ void finalize_bn_kernel(const float* __restrict__ partials,
                                   const float* __restrict__ gamma,
                                   const float* __restrict__ beta,
                                   float* __restrict__ coefA,
                                   float* __restrict__ coefC,
                                   int nblocks, float inv_cnt) {
  __shared__ float tot[256];
  int t = threadIdx.x;
  float s = 0.f;
  for (int b = 0; b < nblocks; ++b) s += partials[(size_t)b * 256 + t];
  tot[t] = s;
  __syncthreads();
  if (t < 128) {
    float mean = tot[t] * inv_cnt;
    float var  = tot[128 + t] * inv_cnt - mean * mean;
    float a = gamma[t] * rsqrtf(var + BN_EPS);
    coefA[t] = a;
    coefC[t] = beta[t] - a * mean;
  }
}

// ---------------------------------------------------------------------------
// BN-apply + ReLU + f32 -> f16, fully vectorized (each element exactly once).
// Each thread converts 8 contiguous elements.
// ---------------------------------------------------------------------------
__global__ void bn_apply_kernel(const float* __restrict__ Z,
                                const float* __restrict__ coefA,
                                const float* __restrict__ coefC,
                                _Float16* __restrict__ X) {
  __shared__ float sa[128], sc[128];
  int tid = threadIdx.x;
  if (tid < 128) { sa[tid] = coefA[tid]; sc[tid] = coefC[tid]; }
  __syncthreads();
  size_t t = (size_t)blockIdx.x * 256 + tid;
  size_t base = t * 8;
  int ch = (int)(base & 127);            // 128 channels per row
  float4 z0 = *(const float4*)(Z + base);
  float4 z1 = *(const float4*)(Z + base + 4);
  v8h o;
  o[0] = (_Float16)fmaxf(sa[ch + 0] * z0.x + sc[ch + 0], 0.f);
  o[1] = (_Float16)fmaxf(sa[ch + 1] * z0.y + sc[ch + 1], 0.f);
  o[2] = (_Float16)fmaxf(sa[ch + 2] * z0.z + sc[ch + 2], 0.f);
  o[3] = (_Float16)fmaxf(sa[ch + 3] * z0.w + sc[ch + 3], 0.f);
  o[4] = (_Float16)fmaxf(sa[ch + 4] * z1.x + sc[ch + 4], 0.f);
  o[5] = (_Float16)fmaxf(sa[ch + 5] * z1.y + sc[ch + 5], 0.f);
  o[6] = (_Float16)fmaxf(sa[ch + 6] * z1.z + sc[ch + 6], 0.f);
  o[7] = (_Float16)fmaxf(sa[ch + 7] * z1.w + sc[ch + 7], 0.f);
  *(v8h*)(X + base) = o;
}

// ---------------------------------------------------------------------------
// three_nn + three_interpolate + concat -> X0 f16 [B*4096][192]
// cols 0..63 = points0, cols 64..191 = inverse-distance blend of points1.
// xyz1 tile (12 KB) staged to LDS with async global->LDS copies when the
// gfx1250 builtin is available (ASYNCcnt path), else plain loads.
// ---------------------------------------------------------------------------
__global__ void knn_interp_kernel(const float* __restrict__ xyz0,
                                  const float* __restrict__ xyz1,
                                  const float* __restrict__ p0,
                                  const float* __restrict__ p1,
                                  _Float16* __restrict__ X0) {
  __shared__ float sx[1024 * 3];          // 12 KB: xyz1 for this batch
  int b = blockIdx.x >> 4;                // 16 chunks of 256 queries per batch
  int chunk = blockIdx.x & 15;
  int tid = threadIdx.x;
  const float* src = xyz1 + (size_t)b * 1024 * 3;
#if defined(ATHENA_ASYNC_LDS)
  // 768 x 16B chunks, 3 per thread, DMA'd past the VGPR file.
  // Builtin signature (from toolchain diagnostic): (v4i global*, v4i local*,
  // int offset, int cpol). Flat pointers to global memory are numerically
  // identical to addrspace(1) addresses, so cast via integer for arg 0.
  #pragma unroll
  for (int r = 0; r < 3; ++r) {
    int t = tid + r * 256;
    v4i __attribute__((address_space(1)))* gsrc =
        (v4i __attribute__((address_space(1)))*)(unsigned long long)(src + t * 4);
    v4i __attribute__((address_space(3)))* ldst =
        (v4i __attribute__((address_space(3)))*)(&sx[t * 4]);
    __builtin_amdgcn_global_load_async_to_lds_b128(gsrc, ldst, 0, 0);
  }
#if __has_builtin(__builtin_amdgcn_s_wait_asynccnt)
  __builtin_amdgcn_s_wait_asynccnt(0);
#else
  asm volatile("s_wait_asynccnt 0x0" ::: "memory");
#endif
#else
  for (int t = tid; t < 1024 * 3; t += 256) sx[t] = src[t];
#endif
  __syncthreads();

  int i = chunk * 256 + tid;
  size_t row = (size_t)b * 4096 + i;
  float qx = xyz0[row * 3 + 0], qy = xyz0[row * 3 + 1], qz = xyz0[row * 3 + 2];

  float d0 = 3.4e38f, d1 = 3.4e38f, d2 = 3.4e38f;
  int   i0 = 0, i1 = 0, i2 = 0;
  for (int j = 0; j < 1024; ++j) {
    float dx = sx[3 * j + 0] - qx;
    float dy = sx[3 * j + 1] - qy;
    float dz = sx[3 * j + 2] - qz;
    float d = dx * dx + dy * dy + dz * dz;
    if (d < d2) {
      if (d < d1) {
        if (d < d0) { d2 = d1; i2 = i1; d1 = d0; i1 = i0; d0 = d; i0 = j; }
        else        { d2 = d1; i2 = i1; d1 = d;  i1 = j; }
      } else        { d2 = d;  i2 = j; }
    }
  }
  d0 = fmaxf(d0, 1e-10f); d1 = fmaxf(d1, 1e-10f); d2 = fmaxf(d2, 1e-10f);
  float w0 = 1.f / d0, w1 = 1.f / d1, w2 = 1.f / d2;
  float wsum = w0 + w1 + w2;
  w0 /= wsum; w1 /= wsum; w2 /= wsum;

  const float* r0 = p1 + ((size_t)b * 1024 + i0) * 128;
  const float* r1 = p1 + ((size_t)b * 1024 + i1) * 128;
  const float* r2 = p1 + ((size_t)b * 1024 + i2) * 128;
  const float* q  = p0 + row * 64;
  _Float16* xo = X0 + row * 192;
  #pragma unroll 8
  for (int c = 0; c < 64; ++c) xo[c] = (_Float16)q[c];
  #pragma unroll 8
  for (int c = 0; c < 128; ++c)
    xo[64 + c] = (_Float16)(w0 * r0[c] + w1 * r1[c] + w2 * r2[c]);
}

// ---------------------------------------------------------------------------
// WMMA GEMM: Z[M][N] = A[M][K] @ W[K][N]   (A f16, Wt passed as [N][K] f16)
//   STATS    : per-block per-channel sum/sumsq written to private partials
//              slice (deterministic, no atomics).
//   BIASRELU : epilogue z = relu(z + bias[n])  (dense head d1).
// Block = (N/16) waves; each wave owns one 16-wide N strip across 8 M-subtiles.
// ---------------------------------------------------------------------------
template <int K, bool STATS, bool BIASRELU>
__global__ void gemm_wmma(const _Float16* __restrict__ A16,
                          const _Float16* __restrict__ Wt,
                          const float* __restrict__ bias,
                          float* __restrict__ Zout,
                          float* __restrict__ partials, int N) {
  constexpr int KT = K / 32;
  int lane = threadIdx.x & 31;
  int wave = threadIdx.x >> 5;
  int n0 = wave * 16;
  int nn = n0 + (lane & 15);
  int hi = lane >> 4;                       // which half-wave

  // B fragments: per-lane column nn, halves h <-> K = kt*32 + hi*16 + h
  v16h bfrag[KT];
  {
    const _Float16* wp = Wt + (size_t)nn * K + hi * 16;
    #pragma unroll
    for (int kt = 0; kt < KT; ++kt) {
      v8h lo = *(const v8h*)(wp + kt * 32);
      v8h hh = *(const v8h*)(wp + kt * 32 + 8);
      v16h bf;
      #pragma unroll
      for (int t = 0; t < 8; ++t) { bf[t] = lo[t]; bf[8 + t] = hh[t]; }
      bfrag[kt] = bf;
    }
  }

  float colsum = 0.f, colsq = 0.f;
  size_t mblk = (size_t)blockIdx.x * 128;
  float biasv = BIASRELU ? bias[nn] : 0.f;

  #pragma unroll 1
  for (int ms = 0; ms < 8; ++ms) {
    size_t m0 = mblk + ms * 16;
    size_t arow = m0 + (lane & 15);
    const _Float16* abase = A16 + arow * K;
    v8f c = {};
    #pragma unroll
    for (int kt = 0; kt < KT; ++kt) {
      int kbase = kt * 32 + hi * 16;        // A layout matches B: halves 0-7 at kbase,
      v16h a;                               // halves 8-15 at kbase+8
      v8h lo = *(const v8h*)(abase + kbase);
      v8h hh = *(const v8h*)(abase + kbase + 8);
      #pragma unroll
      for (int t = 0; t < 8; ++t) { a[t] = lo[t]; a[8 + t] = hh[t]; }
      c = __builtin_amdgcn_wmma_f32_16x16x32_f16(false, a, false, bfrag[kt],
                                                 (short)0, c, false, false);
    }
    // epilogue: D VGPR r -> row m0 + r + hi*8, col nn
    float* zrow = Zout + (m0 + hi * 8) * N + nn;
    #pragma unroll
    for (int r = 0; r < 8; ++r) {
      float v = c[r];
      if (BIASRELU) v = fmaxf(v + biasv, 0.f);
      zrow[(size_t)r * N] = v;
      if (STATS) { colsum += v; colsq += v * v; }
    }
  }
  if (STATS) {
    colsum += __shfl_xor(colsum, 16, 32);   // lanes l and l+16 share column nn
    colsq  += __shfl_xor(colsq, 16, 32);
    if (lane < 16) {                        // block-private slice: no atomics
      float* pb = partials + (size_t)blockIdx.x * 256;
      pb[nn]       = colsum;
      pb[128 + nn] = colsq;
    }
  }
}

// ---------------------------------------------------------------------------
// head tail: h2 = relu(H1 @ d2_w + d2_b); out = tanh(h2 @ d3_w + d3_b)
// ---------------------------------------------------------------------------
__global__ void head_tail_kernel(const float* __restrict__ H1,
                                 const float* __restrict__ w2,
                                 const float* __restrict__ b2,
                                 const float* __restrict__ w3,
                                 const float* __restrict__ b3,
                                 float* __restrict__ out) {
  __shared__ float sw2[64 * 32];
  __shared__ float sb2[32];
  __shared__ float sw3[32 * 3];
  __shared__ float sb3[3];
  int tid = threadIdx.x;
  for (int t = tid; t < 64 * 32; t += 256) sw2[t] = w2[t];
  if (tid < 32) sb2[tid] = b2[tid];
  if (tid < 96) sw3[tid] = w3[tid];
  if (tid < 3)  sb3[tid] = b3[tid];
  __syncthreads();

  size_t row = (size_t)blockIdx.x * 256 + tid;
  const float* h = H1 + row * 64;
  float h1[64];
  #pragma unroll
  for (int k = 0; k < 64; ++k) h1[k] = h[k];
  float h2[32];
  #pragma unroll 4
  for (int c = 0; c < 32; ++c) {
    float acc = sb2[c];
    #pragma unroll
    for (int k = 0; k < 64; ++k) acc += h1[k] * sw2[k * 32 + c];
    h2[c] = fmaxf(acc, 0.f);
  }
  #pragma unroll
  for (int o = 0; o < 3; ++o) {
    float acc = sb3[o];
    #pragma unroll
    for (int k = 0; k < 32; ++k) acc += h2[k] * sw3[k * 3 + o];
    out[row * 3 + o] = tanhf(acc);
  }
}

// ---------------------------------------------------------------------------
// host launch
// ---------------------------------------------------------------------------
extern "C" void kernel_launch(void* const* d_in, const int* in_sizes, int n_in,
                              void* d_out, int out_size, void* d_ws, size_t ws_size,
                              hipStream_t stream) {
  (void)out_size; (void)ws_size;
  const int B = 32, N0 = 4096, N1 = 1024, C0 = 64, C1 = 128;
  const int K1 = 192, H = 128, D1 = 64;
  const int M = B * N0;                       // 131072 rows
  const int NBLK = M / 128;                   // 1024 GEMM blocks

  auto find1 = [&](int sz) -> int {
    for (int i = 0; i < n_in; ++i) if (in_sizes[i] == sz) return i;
    return -1;
  };
  auto find2 = [&](int sz) -> int {
    int c = 0;
    for (int i = 0; i < n_in; ++i)
      if (in_sizes[i] == sz && ++c == 2) return i;
    return -1;
  };

  // Only the fp4 path + dense heads are live (fp1/fp2/fp3 results are dead).
  int i_xyz0 = find1(B * N0 * 3);             // 393216 (unique)
  int i_p0   = find1(B * N0 * C0);            // 8388608 (unique)
  int i_p1   = find1(B * N1 * C1);            // 4194304 (unique)
  // xyz1 (98304) collides with fp3L0.w; disambiguate by adjacency to xyz0.
  int i_xyz1 = (i_xyz0 + 1 < n_in && in_sizes[i_xyz0 + 1] == B * N1 * 3)
                   ? i_xyz0 + 1 : i_xyz0 + 2;

  int i_d3w = find1(32 * 3);                  // 96 (unique)
  bool insertion = (i_d3w + 1 < n_in && in_sizes[i_d3w + 1] == 3);

  int iw0 = find1(K1 * H);                    // fp4 L0 w (24576, unique)
  int iw1 = find1(H * H);                     // fp4 L1 w (first 16384)
  int iw2 = find2(H * H);                     // fp4 L2 w (second 16384)
  auto gb = [&](int iw, int& ig, int& ib) {
    if (insertion) { ig = iw + 2; ib = iw + 3; }   // layer dict {w,b,gamma,beta}
    else           { ig = iw - 1; ib = iw - 2; }   // sorted {b,beta,gamma,w}
  };
  int ig0, ib0, ig1, ib1, ig2, ib2;
  gb(iw0, ig0, ib0); gb(iw1, ig1, ib1); gb(iw2, ig2, ib2);

  int i_d1w = find1(H * D1);                  // 8192 (unique)
  int i_d1b = find1(D1);                      // 64 (unique)
  int i_d2w = find1(D1 * 32);                 // 2048 (unique)
  int i_d2b = find1(32);                      // 32 (unique)
  int i_d3b = find1(3);                       // 3 (unique)

  if (i_xyz0 < 0 || i_p0 < 0 || i_p1 < 0 || iw0 < 0 || iw1 < 0 || iw2 < 0 ||
      i_d1w < 0 || i_d1b < 0 || i_d2w < 0 || i_d2b < 0 || i_d3w < 0 || i_d3b < 0)
    return;

  const float* xyz0 = (const float*)d_in[i_xyz0];
  const float* xyz1 = (const float*)d_in[i_xyz1];
  const float* p0   = (const float*)d_in[i_p0];
  const float* p1   = (const float*)d_in[i_p1];

  // workspace layout (~150 MB total)
  char* ws = (char*)d_ws;
  _Float16* WT1  = (_Float16*)(ws + 0);           // [128][192] f16
  _Float16* WT2  = (_Float16*)(ws + 49152);       // [128][128] f16
  _Float16* WT3  = (_Float16*)(ws + 81920);       // [128][128] f16
  _Float16* WTD1 = (_Float16*)(ws + 114688);      // [ 64][128] f16
  float* coefs   = (float*)(ws + 131072);         // 3 layers x {a[128],c[128]}
  float* a1 = coefs,        *c1 = coefs + 128;
  float* a2 = coefs + 256,  *c2 = coefs + 384;
  float* a3 = coefs + 512,  *c3 = coefs + 640;
  const size_t MB = (size_t)1 << 20;
  float*    P  = (float*)(ws + 2 * MB);           // partials: 1024 x 256 f32
  _Float16* X0 = (_Float16*)(ws + 4 * MB);        // [M][192] f16, 48 MB
  float*    Z  = (float*)(ws + 4 * MB + (size_t)M * K1 * 2);   // [M][128] f32
  _Float16* XA = (_Float16*)(ws + 4 * MB + (size_t)M * K1 * 2
                                        + (size_t)M * H * 4);  // [M][128] f16
  float*    H1 = (float*)X0;  // X0 dead after layer-1 GEMM; reuse for head

  // prep: convert+transpose weights to f16 [N][K]
  transpose_f32_to_f16<<<(K1 * H + 255) / 256, 256, 0, stream>>>(
      (const float*)d_in[iw0], WT1, K1, H);
  transpose_f32_to_f16<<<(H * H + 255) / 256, 256, 0, stream>>>(
      (const float*)d_in[iw1], WT2, H, H);
  transpose_f32_to_f16<<<(H * H + 255) / 256, 256, 0, stream>>>(
      (const float*)d_in[iw2], WT3, H, H);
  transpose_f32_to_f16<<<(H * D1 + 255) / 256, 256, 0, stream>>>(
      (const float*)d_in[i_d1w], WTD1, H, D1);

  // three_nn + interpolate + concat -> X0 f16 [M][192]
  knn_interp_kernel<<<B * 16, 256, 0, stream>>>(xyz0, xyz1, p0, p1, X0);

  const float inv_cnt = 1.f / (float)M;
  const int APPLY_BLKS = (M * H / 8) / 256;       // 8192

  // layer 1: Z = X0 @ W1 (bias folds out of BN); deterministic stats
  gemm_wmma<192, true, false><<<NBLK, 256, 0, stream>>>(X0, WT1, nullptr, Z, P, 128);
  finalize_bn_kernel<<<1, 256, 0, stream>>>(
      P, (const float*)d_in[ig0], (const float*)d_in[ib0], a1, c1, NBLK, inv_cnt);
  bn_apply_kernel<<<APPLY_BLKS, 256, 0, stream>>>(Z, a1, c1, XA);

  // layer 2
  gemm_wmma<128, true, false><<<NBLK, 256, 0, stream>>>(XA, WT2, nullptr, Z, P, 128);
  finalize_bn_kernel<<<1, 256, 0, stream>>>(
      P, (const float*)d_in[ig1], (const float*)d_in[ib1], a2, c2, NBLK, inv_cnt);
  bn_apply_kernel<<<APPLY_BLKS, 256, 0, stream>>>(Z, a2, c2, XA);

  // layer 3
  gemm_wmma<128, true, false><<<NBLK, 256, 0, stream>>>(XA, WT3, nullptr, Z, P, 128);
  finalize_bn_kernel<<<1, 256, 0, stream>>>(
      P, (const float*)d_in[ig2], (const float*)d_in[ib2], a3, c3, NBLK, inv_cnt);
  bn_apply_kernel<<<APPLY_BLKS, 256, 0, stream>>>(Z, a3, c3, XA);

  // head d1: H1 = relu(XA @ Wd1 + b_d1),  N=64 -> 4 waves/block
  gemm_wmma<128, false, true><<<NBLK, 128, 0, stream>>>(
      XA, WTD1, (const float*)d_in[i_d1b], H1, nullptr, 64);

  // head d2/d3: 64 -> 32 (relu) -> 3 (tanh)
  head_tail_kernel<<<M / 256, 256, 0, stream>>>(
      H1, (const float*)d_in[i_d2w], (const float*)d_in[i_d2b],
      (const float*)d_in[i_d3w], (const float*)d_in[i_d3b], (float*)d_out);
}
